// DecoderLayer_63720134804169
// MI455X (gfx1250) — compile-verified, matrix operand order checked by reading, not run on hardware
//
#include <hip/hip_runtime.h>
#include <type_traits>

// ---------------- problem constants ----------------
constexpr int B_   = 4;
constexpr int NQ_  = 2500;
constexpr int C_   = 256;
constexpr int H_   = 8;
constexpr int P_   = 8;
constexpr int DFF_ = 512;
constexpr int DH_  = 32;
constexpr int BEV  = 200;
constexpr int NBEV = BEV * BEV;        // 40000
constexpr int M1   = B_ * NQ_;         // 10000 (multiple of 16)
constexpr int MV   = B_ * NBEV;        // 160000 (multiple of 16)
constexpr int QT_  = (NQ_ + 15) / 16;  // 157 query tiles
constexpr int KBLK = (NQ_ + 31) / 32;  // 79 key blocks of 32

typedef __attribute__((ext_vector_type(16))) _Float16 v16h;
typedef __attribute__((ext_vector_type(8)))  float    v8f;
typedef __attribute__((ext_vector_type(8)))  _Float16 h8;
typedef __attribute__((ext_vector_type(4)))  float    f4;

// ---------------- f32 -> f16 weight conversion ----------------
__global__ __launch_bounds__(256) void cvt_f32_f16(const float* __restrict__ x,
                                                   _Float16* __restrict__ y, int n) {
  int i = blockIdx.x * 256 + threadIdx.x;
  if (i < n) y[i] = (_Float16)x[i];
}

// ---------------- generic WMMA GEMM: D = act(A @ W^T + bias [+resid]) ----------------
// A: [M,K] (f32 or f16), W: [N,K] f16 (torch Linear layout), bias: [N] f32,
// resid: [M,N] f32 (optional), D: [M,N] (f32 or f16).
// M%16==0, N%64==0, K%32==0. Each wave computes a 16x64 strip: one A fragment
// feeds 4 WMMAs per k-step (4x less activation re-fetch than 16x16 tiling).
template <typename AT, typename OT, bool ADD2, bool RELU, bool RES>
__global__ __launch_bounds__(256) void gemm_wmma(const AT* __restrict__ A,
                                                 const float* __restrict__ A2,
                                                 const _Float16* __restrict__ W,
                                                 const float* __restrict__ bias,
                                                 const float* __restrict__ resid,
                                                 OT* __restrict__ D,
                                                 int M, int N, int K) {
  const int wid  = threadIdx.x >> 5;
  const int lane = threadIdx.x & 31;
  const int ntg   = N >> 6;                   // 64-wide n-groups
  const int total = (M >> 4) * ntg;
  const int tile = blockIdx.x * 8 + wid;
  if (tile >= total) return;                  // wave-uniform exit
  const int mt = tile / ntg;
  const int ng = tile - mt * ntg;
  const int lid = lane & 15;
  const int kb  = (lane >> 4) * 8;

  const size_t arow  = (size_t)(mt * 16 + lid) * K;
  const size_t wbase = (size_t)(ng * 64 + lid) * K;
  const size_t wstep = (size_t)16 * K;

  v8f acc0 = {}, acc1 = {}, acc2 = {}, acc3 = {};
  for (int ks = 0; ks < K; ks += 32) {
    // A fragment (shared by 4 WMMAs)
    v16h af;
    if constexpr (std::is_same<AT, float>::value) {
      const float* ap = A + arow + ks;
      f4 x0 = *(const f4*)(ap + kb);
      f4 x1 = *(const f4*)(ap + kb + 4);
      f4 x2 = *(const f4*)(ap + kb + 16);
      f4 x3 = *(const f4*)(ap + kb + 20);
      if constexpr (ADD2) {
        const float* ap2 = A2 + arow + ks;
        x0 += *(const f4*)(ap2 + kb);
        x1 += *(const f4*)(ap2 + kb + 4);
        x2 += *(const f4*)(ap2 + kb + 16);
        x3 += *(const f4*)(ap2 + kb + 20);
      }
#pragma unroll
      for (int i = 0; i < 4; ++i) {
        af[i]      = (_Float16)x0[i];
        af[4 + i]  = (_Float16)x1[i];
        af[8 + i]  = (_Float16)x2[i];
        af[12 + i] = (_Float16)x3[i];
      }
    } else {
      const _Float16* ap = A + arow + ks + kb;
      h8 a0 = *(const h8*)ap;
      h8 a1 = *(const h8*)(ap + 16);
#pragma unroll
      for (int i = 0; i < 8; ++i) { af[i] = a0[i]; af[8 + i] = a1[i]; }
    }

    // 4 B fragments from f16 weights (contiguous along K)
    v16h wf[4];
#pragma unroll
    for (int t = 0; t < 4; ++t) {
      const _Float16* wp = W + wbase + (size_t)t * wstep + ks + kb;
      h8 w0 = *(const h8*)wp;
      h8 w1 = *(const h8*)(wp + 16);
#pragma unroll
      for (int i = 0; i < 8; ++i) { wf[t][i] = w0[i]; wf[t][8 + i] = w1[i]; }
    }

    acc0 = __builtin_amdgcn_wmma_f32_16x16x32_f16(false, af, false, wf[0], (short)0, acc0, false, false);
    acc1 = __builtin_amdgcn_wmma_f32_16x16x32_f16(false, af, false, wf[1], (short)0, acc1, false, false);
    acc2 = __builtin_amdgcn_wmma_f32_16x16x32_f16(false, af, false, wf[2], (short)0, acc2, false, false);
    acc3 = __builtin_amdgcn_wmma_f32_16x16x32_f16(false, af, false, wf[3], (short)0, acc3, false, false);
  }

#pragma unroll
  for (int t = 0; t < 4; ++t) {
    const v8f acc = (t == 0) ? acc0 : (t == 1) ? acc1 : (t == 2) ? acc2 : acc3;
    const int ncol = ng * 64 + t * 16 + lid;
    const float bn = bias[ncol];
#pragma unroll
    for (int j = 0; j < 8; ++j) {
      const int m = mt * 16 + (lane >> 4) * 8 + j;
      const size_t o = (size_t)m * N + ncol;
      float v = acc[j] + bn;
      if constexpr (RELU) v = fmaxf(v, 0.f);
      if constexpr (RES)  v += resid[o];
      D[o] = (OT)v;
    }
  }
}

// ---------------- flash self-attention (one 16-row query tile per wave) ----------------
__global__ __launch_bounds__(256) void flash_attn_kernel(const _Float16* __restrict__ qh,
                                                         const _Float16* __restrict__ kh,
                                                         const _Float16* __restrict__ vh,
                                                         _Float16* __restrict__ sah) {
  __shared__ float    s_sc[8][16][32];   // raw score tile
  __shared__ _Float16 s_p [8][16][32];   // exp(P) tile (A fragment source)
  __shared__ _Float16 s_v [8][32][32];   // staged V block (key x dh)
  __shared__ float    s_max[8][16];
  __shared__ float    s_sum[8][16];
  __shared__ float    s_corr[8][16];

  const int wid  = threadIdx.x >> 5;
  const int lane = threadIdx.x & 31;
  const int task = blockIdx.x * 8 + wid;
  if (task >= B_ * H_ * QT_) return;
  const int qt = task % QT_;
  const int h  = (task / QT_) & 7;
  const int b  = task / (QT_ * H_);

  const int hf  = lane >> 4;   // 0 or 1
  const int lid = lane & 15;
  const int kb  = hf * 8;

  // Q fragment (A layout), zero-padded past NQ
  v16h aq;
  {
    const int m = qt * 16 + lid;
    if (m < NQ_) {
      const _Float16* qp = qh + ((size_t)b * NQ_ + m) * C_ + h * DH_ + kb;
      h8 q0 = *(const h8*)qp;
      h8 q1 = *(const h8*)(qp + 16);
#pragma unroll
      for (int i = 0; i < 8; ++i) { aq[i] = q0[i]; aq[8 + i] = q1[i]; }
    } else {
#pragma unroll
      for (int i = 0; i < 16; ++i) aq[i] = (_Float16)0.f;
    }
  }

  v8f acc0 = {};
  v8f acc1 = {};
  if (lane < 16) { s_max[wid][lane] = -1e30f; s_sum[wid][lane] = 0.f; }
  asm volatile("s_wait_dscnt 0" ::: "memory");

  const float scale = 0.17677669529663687f;  // 1/sqrt(32)

  for (int kbl = 0; kbl < KBLK; ++kbl) {
    const int key0 = kbl * 32;

    // stage V block: lane -> one key row of 32 halves
    {
      const int key = key0 + lane;
      _Float16* dst = &s_v[wid][lane][0];
      if (key < NQ_) {
        const _Float16* vp = vh + ((size_t)b * NQ_ + key) * C_ + h * DH_;
        *(h8*)(dst)      = *(const h8*)(vp);
        *(h8*)(dst + 8)  = *(const h8*)(vp + 8);
        *(h8*)(dst + 16) = *(const h8*)(vp + 16);
        *(h8*)(dst + 24) = *(const h8*)(vp + 24);
      } else {
        h8 z = {};
        *(h8*)(dst) = z; *(h8*)(dst + 8) = z; *(h8*)(dst + 16) = z; *(h8*)(dst + 24) = z;
      }
    }

    // two 16-key score subtiles: S = Q x K^T
#pragma unroll
    for (int t = 0; t < 2; ++t) {
      const int key = key0 + t * 16 + lid;
      v16h bk;
      if (key < NQ_) {
        const _Float16* kp = kh + ((size_t)b * NQ_ + key) * C_ + h * DH_ + kb;
        h8 k0 = *(const h8*)kp;
        h8 k1 = *(const h8*)(kp + 16);
#pragma unroll
        for (int i = 0; i < 8; ++i) { bk[i] = k0[i]; bk[8 + i] = k1[i]; }
      } else {
#pragma unroll
        for (int i = 0; i < 16; ++i) bk[i] = (_Float16)0.f;
      }
      v8f z = {};
      v8f s = __builtin_amdgcn_wmma_f32_16x16x32_f16(false, aq, false, bk,
                                                     (short)0, z, false, false);
      const bool kvalid = (key < NQ_);
#pragma unroll
      for (int j = 0; j < 8; ++j) {
        const int m = hf * 8 + j;
        s_sc[wid][m][t * 16 + lid] = kvalid ? s[j] * scale : -1e30f;
      }
    }
    asm volatile("s_wait_dscnt 0" ::: "memory");

    // online softmax update: lanes 0..15 each own one row
    if (lane < 16) {
      const int r = lane;
      const float om = s_max[wid][r];
      float tm = om;
#pragma unroll
      for (int j = 0; j < 32; ++j) tm = fmaxf(tm, s_sc[wid][r][j]);
      const float corr = __expf(om - tm);
      float sum = s_sum[wid][r] * corr;
#pragma unroll
      for (int j = 0; j < 32; ++j) {
        const float e = __expf(s_sc[wid][r][j] - tm);
        sum += e;
        s_p[wid][r][j] = (_Float16)e;
      }
      s_max[wid][r] = tm;
      s_sum[wid][r] = sum;
      s_corr[wid][r] = corr;
    }
    asm volatile("s_wait_dscnt 0" ::: "memory");

    // rescale accumulators by per-row correction
#pragma unroll
    for (int j = 0; j < 8; ++j) {
      const float f = s_corr[wid][hf * 8 + j];
      acc0[j] *= f;
      acc1[j] *= f;
    }

    // P fragment (A layout) from LDS
    v16h ap;
    {
      const _Float16* pp = &s_p[wid][lid][kb];
      h8 p0 = *(const h8*)pp;
      h8 p1 = *(const h8*)(pp + 16);
#pragma unroll
      for (int i = 0; i < 8; ++i) { ap[i] = p0[i]; ap[8 + i] = p1[i]; }
    }
    // V fragments (B layout) from staged LDS block
    v16h bv0, bv1;
#pragma unroll
    for (int i = 0; i < 8; ++i) {
      bv0[i]     = s_v[wid][kb + i][lid];
      bv0[8 + i] = s_v[wid][16 + kb + i][lid];
      bv1[i]     = s_v[wid][kb + i][16 + lid];
      bv1[8 + i] = s_v[wid][16 + kb + i][16 + lid];
    }
    acc0 = __builtin_amdgcn_wmma_f32_16x16x32_f16(false, ap, false, bv0,
                                                  (short)0, acc0, false, false);
    acc1 = __builtin_amdgcn_wmma_f32_16x16x32_f16(false, ap, false, bv1,
                                                  (short)0, acc1, false, false);
    asm volatile("s_wait_dscnt 0" ::: "memory");
  }

  // epilogue: divide by row sums, store f16
#pragma unroll
  for (int j = 0; j < 8; ++j) {
    const int mr = hf * 8 + j;
    const int m = qt * 16 + mr;
    if (m < NQ_) {
      const float inv = 1.f / s_sum[wid][mr];
      _Float16* op = sah + ((size_t)b * NQ_ + m) * C_ + h * DH_;
      op[lid]      = (_Float16)(acc0[j] * inv);
      op[16 + lid] = (_Float16)(acc1[j] * inv);
    }
  }
}

// ---------------- LayerNorm: one wave per 256-wide row ----------------
__global__ __launch_bounds__(256) void layernorm_kernel(const float* __restrict__ X,
                                                        const float* __restrict__ g,
                                                        const float* __restrict__ bt,
                                                        float* __restrict__ Y, int rows) {
  const int wid  = threadIdx.x >> 5;
  const int lane = threadIdx.x & 31;
  const int row = blockIdx.x * 8 + wid;
  if (row >= rows) return;
  const float* xp = X + (size_t)row * C_ + lane * 8;
  f4 x0 = *(const f4*)xp;
  f4 x1 = *(const f4*)(xp + 4);
  float s = 0.f, s2 = 0.f;
#pragma unroll
  for (int i = 0; i < 4; ++i) {
    s += x0[i] + x1[i];
    s2 += x0[i] * x0[i] + x1[i] * x1[i];
  }
#pragma unroll
  for (int m = 16; m >= 1; m >>= 1) {
    s  += __shfl_xor(s, m, 32);
    s2 += __shfl_xor(s2, m, 32);
  }
  const float mean = s * (1.f / 256.f);
  const float var  = s2 * (1.f / 256.f) - mean * mean;
  const float inv  = rsqrtf(var + 1e-5f);
  const float* gp = g + lane * 8;
  const float* bp = bt + lane * 8;
  float* yp = Y + (size_t)row * C_ + lane * 8;
#pragma unroll
  for (int i = 0; i < 4; ++i) {
    yp[i]     = (x0[i] - mean) * inv * gp[i]     + bp[i];
    yp[4 + i] = (x1[i] - mean) * inv * gp[4 + i] + bp[4 + i];
  }
}

// ---------------- softmax over P=8 attention weights, in place ----------------
__global__ __launch_bounds__(256) void softmax_aw_kernel(float* __restrict__ aw, int ngroups) {
  const int t = blockIdx.x * 256 + threadIdx.x;
  if (t >= ngroups) return;
  float* p = aw + (size_t)t * P_;
  float m = -1e30f;
#pragma unroll
  for (int i = 0; i < P_; ++i) m = fmaxf(m, p[i]);
  float e[P_];
  float s = 0.f;
#pragma unroll
  for (int i = 0; i < P_; ++i) { e[i] = __expf(p[i] - m); s += e[i]; }
  const float inv = 1.f / s;
#pragma unroll
  for (int i = 0; i < P_; ++i) p[i] = e[i] * inv;
}

// ---------------- deformable bilinear sampling: one wave per (b,q,h), lane=dh ----------------
__global__ __launch_bounds__(256) void deform_sample_kernel(const _Float16* __restrict__ valp,
                                                            const float* __restrict__ offb,
                                                            const float* __restrict__ awb,
                                                            const float* __restrict__ ref2d,
                                                            _Float16* __restrict__ outp) {
  const int wid  = threadIdx.x >> 5;
  const int lane = threadIdx.x & 31;   // dh index
  const int task = blockIdx.x * 8 + wid;
  if (task >= M1 * H_) return;
  const int rowq = task >> 3;
  const int h    = task & 7;
  const int b    = rowq / NQ_;

  const float rx = ref2d[(size_t)rowq * 2 + 0];
  const float ry = ref2d[(size_t)rowq * 2 + 1];
  const size_t vbase = (size_t)b * NBEV * C_ + h * DH_ + lane;

  float acc = 0.f;
#pragma unroll
  for (int p = 0; p < P_; ++p) {
    const float ox = offb[(size_t)rowq * (H_ * P_ * 2) + (h * P_ + p) * 2 + 0];
    const float oy = offb[(size_t)rowq * (H_ * P_ * 2) + (h * P_ + p) * 2 + 1];
    const float a  = awb[(size_t)rowq * (H_ * P_) + h * P_ + p];
    const float x = rx * (float)BEV + ox - 0.5f;
    const float y = ry * (float)BEV + oy - 0.5f;
    const float xf = floorf(x), yf = floorf(y);
    const float lx = x - xf, ly = y - yf;
    const int xi = (int)xf, yi = (int)yf;
    const float w00 = (1.f - lx) * (1.f - ly);
    const float w01 = lx * (1.f - ly);
    const float w10 = (1.f - lx) * ly;
    const float w11 = lx * ly;
    float samp = 0.f;
#pragma unroll
    for (int c = 0; c < 4; ++c) {
      const int cx = xi + (c & 1);
      const int cy = yi + (c >> 1);
      const float w = (c == 0) ? w00 : (c == 1) ? w01 : (c == 2) ? w10 : w11;
      if (cx >= 0 && cx < BEV && cy >= 0 && cy < BEV) {
        const float gval = (float)valp[vbase + (size_t)(cy * BEV + cx) * C_];
        samp += w * gval;
      }
    }
    acc += a * samp;
  }
  outp[(size_t)rowq * C_ + h * DH_ + lane] = (_Float16)acc;
}

// ---------------- launch ----------------
extern "C" void kernel_launch(void* const* d_in, const int* in_sizes, int n_in,
                              void* d_out, int out_size, void* d_ws, size_t ws_size,
                              hipStream_t stream) {
  const float* query      = (const float*)d_in[0];
  const float* value      = (const float*)d_in[1];
  const float* query_pos  = (const float*)d_in[2];
  const float* ref2d      = (const float*)d_in[3];
  const float* in_proj_w  = (const float*)d_in[4];
  const float* in_proj_b  = (const float*)d_in[5];
  const float* out_proj_w = (const float*)d_in[6];
  const float* out_proj_b = (const float*)d_in[7];
  const float* off_w      = (const float*)d_in[8];
  const float* off_b      = (const float*)d_in[9];
  const float* aw_w       = (const float*)d_in[10];
  const float* aw_b       = (const float*)d_in[11];
  const float* vproj_w    = (const float*)d_in[12];
  const float* vproj_b    = (const float*)d_in[13];
  const float* oproj_w    = (const float*)d_in[14];
  const float* oproj_b    = (const float*)d_in[15];
  const float* ffn_w1     = (const float*)d_in[16];
  const float* ffn_b1     = (const float*)d_in[17];
  const float* ffn_w2     = (const float*)d_in[18];
  const float* ffn_b2     = (const float*)d_in[19];
  const float* ln1_g = (const float*)d_in[20];
  const float* ln1_b = (const float*)d_in[21];
  const float* ln2_g = (const float*)d_in[22];
  const float* ln2_b = (const float*)d_in[23];
  const float* ln3_g = (const float*)d_in[24];
  const float* ln3_b = (const float*)d_in[25];

  char* ws = (char*)d_ws;
  size_t cur = 0;
  auto alloc = [&](size_t bytes) -> void* {
    void* p = ws + cur;
    cur = (cur + bytes + 255) & ~(size_t)255;
    return p;
  };

  // f16 weights
  _Float16* inproj_h = (_Float16*)alloc(3 * C_ * C_ * 2);
  _Float16* outw_h   = (_Float16*)alloc(C_ * C_ * 2);
  _Float16* offw_h   = (_Float16*)alloc(H_ * P_ * 2 * C_ * 2);
  _Float16* aww_h    = (_Float16*)alloc(H_ * P_ * C_ * 2);
  _Float16* vprojw_h = (_Float16*)alloc(C_ * C_ * 2);
  _Float16* oprojw_h = (_Float16*)alloc(C_ * C_ * 2);
  _Float16* ffn1w_h  = (_Float16*)alloc(DFF_ * C_ * 2);
  _Float16* ffn2w_h  = (_Float16*)alloc(C_ * DFF_ * 2);
  // activations
  _Float16* q_h      = (_Float16*)alloc((size_t)M1 * C_ * 2);
  _Float16* k_h      = (_Float16*)alloc((size_t)M1 * C_ * 2);
  _Float16* v_h      = (_Float16*)alloc((size_t)M1 * C_ * 2);
  _Float16* sa_h     = (_Float16*)alloc((size_t)M1 * C_ * 2);
  float*    x0       = (float*)alloc((size_t)M1 * C_ * 4);
  float*    xln1     = (float*)alloc((size_t)M1 * C_ * 4);
  _Float16* valp_h   = (_Float16*)alloc((size_t)MV * C_ * 2);   // 82 MB, L2-resident
  float*    offbuf   = (float*)alloc((size_t)M1 * H_ * P_ * 2 * 4);
  float*    awbuf    = (float*)alloc((size_t)M1 * H_ * P_ * 4);
  _Float16* samp_h   = (_Float16*)alloc((size_t)M1 * C_ * 2);
  float*    x1       = (float*)alloc((size_t)M1 * C_ * 4);
  float*    x2       = (float*)alloc((size_t)M1 * C_ * 4);
  _Float16* ffnh_h   = (_Float16*)alloc((size_t)M1 * DFF_ * 2);
  float*    x3       = (float*)alloc((size_t)M1 * C_ * 4);

  auto cvt = [&](const float* src, _Float16* dst, int n) {
    cvt_f32_f16<<<(n + 255) / 256, 256, 0, stream>>>(src, dst, n);
  };
  cvt(in_proj_w,  inproj_h, 3 * C_ * C_);
  cvt(out_proj_w, outw_h,   C_ * C_);
  cvt(off_w,      offw_h,   H_ * P_ * 2 * C_);
  cvt(aw_w,       aww_h,    H_ * P_ * C_);
  cvt(vproj_w,    vprojw_h, C_ * C_);
  cvt(oproj_w,    oprojw_h, C_ * C_);
  cvt(ffn_w1,     ffn1w_h,  DFF_ * C_);
  cvt(ffn_w2,     ffn2w_h,  C_ * DFF_);

  auto blocks_for = [](int M, int N) { return ((M >> 4) * (N >> 6) + 7) / 8; };

  // QKV projections (q adds query_pos to A)
  gemm_wmma<float, _Float16, true,  false, false><<<blocks_for(M1, C_), 256, 0, stream>>>(
      query, query_pos, inproj_h,              in_proj_b,          nullptr, q_h, M1, C_, C_);
  gemm_wmma<float, _Float16, false, false, false><<<blocks_for(M1, C_), 256, 0, stream>>>(
      query, nullptr,   inproj_h + C_ * C_,    in_proj_b + C_,     nullptr, k_h, M1, C_, C_);
  gemm_wmma<float, _Float16, false, false, false><<<blocks_for(M1, C_), 256, 0, stream>>>(
      query, nullptr,   inproj_h + 2 * C_ * C_, in_proj_b + 2 * C_, nullptr, v_h, M1, C_, C_);

  // fused flash self-attention
  flash_attn_kernel<<<(B_ * H_ * QT_) / 8, 256, 0, stream>>>(q_h, k_h, v_h, sa_h);

  // out-proj + residual(query) -> x0 ; LN1 -> xln1
  gemm_wmma<_Float16, float, false, false, true><<<blocks_for(M1, C_), 256, 0, stream>>>(
      sa_h, nullptr, outw_h, out_proj_b, query, x0, M1, C_, C_);
  layernorm_kernel<<<M1 / 8, 256, 0, stream>>>(x0, ln1_g, ln1_b, xln1, M1);

  // value projection -> f16 (L2-resident)
  gemm_wmma<float, _Float16, false, false, false><<<blocks_for(MV, C_), 256, 0, stream>>>(
      value, nullptr, vprojw_h, vproj_b, nullptr, valp_h, MV, C_, C_);

  // offsets + attention-weight logits
  gemm_wmma<float, float, false, false, false><<<blocks_for(M1, H_ * P_ * 2), 256, 0, stream>>>(
      xln1, nullptr, offw_h, off_b, nullptr, offbuf, M1, H_ * P_ * 2, C_);
  gemm_wmma<float, float, false, false, false><<<blocks_for(M1, H_ * P_), 256, 0, stream>>>(
      xln1, nullptr, aww_h, aw_b, nullptr, awbuf, M1, H_ * P_, C_);
  softmax_aw_kernel<<<(M1 * H_ + 255) / 256, 256, 0, stream>>>(awbuf, M1 * H_);

  // bilinear sampling
  deform_sample_kernel<<<(M1 * H_ + 7) / 8, 256, 0, stream>>>(valp_h, offbuf, awbuf, ref2d, samp_h);

  // oproj + residual(x0) -> x1 ; LN2 -> x2
  gemm_wmma<_Float16, float, false, false, true><<<blocks_for(M1, C_), 256, 0, stream>>>(
      samp_h, nullptr, oprojw_h, oproj_b, x0, x1, M1, C_, C_);
  layernorm_kernel<<<M1 / 8, 256, 0, stream>>>(x1, ln2_g, ln2_b, x2, M1);

  // FFN: relu(x2 @ W1^T + b1) @ W2^T + b2 + x2 -> x3 ; LN3 -> out
  gemm_wmma<float, _Float16, false, true, false><<<blocks_for(M1, DFF_), 256, 0, stream>>>(
      x2, nullptr, ffn1w_h, ffn_b1, nullptr, ffnh_h, M1, DFF_, C_);
  gemm_wmma<_Float16, float, false, false, true><<<blocks_for(M1, C_), 256, 0, stream>>>(
      ffnh_h, nullptr, ffn2w_h, ffn_b2, x2, x3, M1, C_, DFF_);
  layernorm_kernel<<<M1 / 8, 256, 0, stream>>>(x3, ln3_g, ln3_b, (float*)d_out, M1);
}
